// MLGL_MP_56839597195495
// MI455X (gfx1250) — matrix operand order repaired and checked by reading, not compiled, not run on hardware
//
#include <hip/hip_runtime.h>
#include <hip/hip_bf16.h>
#include <math.h>

#define NN      30000
#define NE      300000
#define NG      64
#define NC      100
#define FXD     78
#define HEADS   10
#define HID     780       // 78*10
#define INCH    300
#define ETOT    (NE + NN) // edges + self loops
#define FC1     1500
#define FC2     1024

typedef __attribute__((ext_vector_type(16))) __bf16 v16bf;
typedef __attribute__((ext_vector_type(8)))  __bf16 v8bf;
typedef __attribute__((ext_vector_type(8)))  float  v8f;

// ---------------- helpers ----------------
__device__ __forceinline__ __bf16 f2bf(float f) {
  unsigned u = __float_as_uint(f);
  u += 0x7fffu + ((u >> 16) & 1u);               // round-to-nearest-even
  unsigned short hs = (unsigned short)(u >> 16);
  __bf16 b; __builtin_memcpy(&b, &hs, 2); return b;
}
// order-preserving float -> uint key (for atomicMax-based segment max)
__device__ __forceinline__ unsigned fkey(float f) {
  unsigned u = __float_as_uint(f);
  return (u & 0x80000000u) ? ~u : (u | 0x80000000u);
}
__device__ __forceinline__ float funkey(unsigned k) {
  unsigned u = (k & 0x80000000u) ? (k & 0x7fffffffu) : ~k;
  return __uint_as_float(u);
}
__device__ __forceinline__ float lrelu2(float x) { return x > 0.f ? x : 0.2f * x; }
__device__ __forceinline__ void edge_sd(int e, const int* __restrict__ ei, int& s, int& d) {
  if (e < NE) { s = ei[e]; d = ei[NE + e]; } else { s = d = e - NE; }
}

// ---------------- bf16 staging (pad + convert + optional transpose) ----------------
__global__ void cvtA_k(const float* __restrict__ A, __bf16* __restrict__ Ab,
                       int M, int K, int lda, int Kp, long long n) {
  long long i = (long long)blockIdx.x * blockDim.x + threadIdx.x;
  if (i >= n) return;
  int m = (int)(i / Kp), k = (int)(i % Kp);
  float v = (m < M && k < K) ? A[(size_t)m * lda + k] : 0.f;
  Ab[i] = f2bf(v);
}
// Bt[n][k] = B[k][n] (Np x Kp), zero padded.
__global__ void cvtBT_k(const float* __restrict__ B, __bf16* __restrict__ Bt,
                        int K, int N, int ldb, int Kp, long long n) {
  long long i = (long long)blockIdx.x * blockDim.x + threadIdx.x;
  if (i >= n) return;
  int nn = (int)(i / Kp), k = (int)(i % Kp);
  float v = (nn < N && k < K) ? B[(size_t)k * ldb + nn] : 0.f;
  Bt[i] = f2bf(v);
}

// ---------------- WMMA bf16 GEMM, 2x2 register-blocked ----------------
// C[M,N] = epi(Ab[Mp,Kp] * Bt[Np,Kp]^T + bias); Mp,Np multiples of 32, Kp of 32.
// One 32x32 macro-tile per wave (4 WMMA tiles), 4 waves/block stacked along M.
// Inner loop: 8x aligned b128 loads -> 4x v_wmma (16 FLOP/byte from L2).
template<int EPI>
__global__ void gemm_wmma_bf16(const __bf16* __restrict__ Ab, const __bf16* __restrict__ Bt,
                               const float* __restrict__ bias, float* __restrict__ C,
                               int M, int N, int Kp, int ldc) {
  const int lane = threadIdx.x & 31;
  const int wave = threadIdx.x >> 5;
  const int m0 = (blockIdx.y * 4 + wave) * 32;
  const int n0 = blockIdx.x * 32;
  if (m0 >= M) return;                        // wave-uniform
  const int lgrp = lane >> 4;                 // 0: lanes 0-15, 1: lanes 16-31
  const int lmod = lane & 15;
  // A fragments (16x32 each, ISA 7.12.2): runs [k0+8g..+7], [k0+16+8g..+7] of a row
  const __bf16* ap0 = Ab + (size_t)(m0 + lmod) * Kp + lgrp * 8;
  const __bf16* ap1 = ap0 + (size_t)16 * Kp;
  // B fragments (32x16 each): runs [k0+8g..+7], [k0+16+8g..+7] of a Bt row
  const __bf16* bp0 = Bt + (size_t)(n0 + lmod) * Kp + lgrp * 8;
  const __bf16* bp1 = bp0 + (size_t)16 * Kp;
  v8f acc00 = {}, acc01 = {}, acc10 = {}, acc11 = {};
  for (int k0 = 0; k0 < Kp; k0 += 32) {
    __builtin_prefetch(ap0 + k0 + 256, 0, 1); // global_prefetch_b8, speculative
    __builtin_prefetch(ap1 + k0 + 256, 0, 1);
    __builtin_prefetch(bp0 + k0 + 256, 0, 1);
    __builtin_prefetch(bp1 + k0 + 256, 0, 1);
    v8bf a0l = *(const v8bf*)(ap0 + k0);
    v8bf a0h = *(const v8bf*)(ap0 + k0 + 16);
    v8bf a1l = *(const v8bf*)(ap1 + k0);
    v8bf a1h = *(const v8bf*)(ap1 + k0 + 16);
    v8bf b0l = *(const v8bf*)(bp0 + k0);
    v8bf b0h = *(const v8bf*)(bp0 + k0 + 16);
    v8bf b1l = *(const v8bf*)(bp1 + k0);
    v8bf b1h = *(const v8bf*)(bp1 + k0 + 16);
    v16bf a0 = __builtin_shufflevector(a0l, a0h, 0,1,2,3,4,5,6,7,8,9,10,11,12,13,14,15);
    v16bf a1 = __builtin_shufflevector(a1l, a1h, 0,1,2,3,4,5,6,7,8,9,10,11,12,13,14,15);
    v16bf b0 = __builtin_shufflevector(b0l, b0h, 0,1,2,3,4,5,6,7,8,9,10,11,12,13,14,15);
    v16bf b1 = __builtin_shufflevector(b1l, b1h, 0,1,2,3,4,5,6,7,8,9,10,11,12,13,14,15);
    acc00 = __builtin_amdgcn_wmma_f32_16x16x32_bf16(false, a0, false, b0, (short)0, acc00, false, false);
    acc01 = __builtin_amdgcn_wmma_f32_16x16x32_bf16(false, a0, false, b1, (short)0, acc01, false, false);
    acc10 = __builtin_amdgcn_wmma_f32_16x16x32_bf16(false, a1, false, b0, (short)0, acc10, false, false);
    acc11 = __builtin_amdgcn_wmma_f32_16x16x32_bf16(false, a1, false, b1, (short)0, acc11, false, false);
  }
  // epilogue: C layout per tile: VGPR r -> row r (+8 for lanes 16-31), col = lane%16
  const int bn0 = n0 + lmod, bn1 = n0 + 16 + lmod;
  float bb0 = (bias && bn0 < N) ? bias[bn0] : 0.f;
  float bb1 = (bias && bn1 < N) ? bias[bn1] : 0.f;
  #pragma unroll
  for (int r = 0; r < 8; ++r) {
    int ma = m0 + r + (lgrp ? 8 : 0);
    int mb = ma + 16;
    if (ma < M) {
      if (bn0 < N) {
        float v = acc00[r] + bb0;
        if (EPI == 1) v = v > 0.f ? v : 0.f;
        if (EPI == 2) v = lrelu2(v);
        C[(size_t)ma * ldc + bn0] = v;
      }
      if (bn1 < N) {
        float v = acc01[r] + bb1;
        if (EPI == 1) v = v > 0.f ? v : 0.f;
        if (EPI == 2) v = lrelu2(v);
        C[(size_t)ma * ldc + bn1] = v;
      }
    }
    if (mb < M) {
      if (bn0 < N) {
        float v = acc10[r] + bb0;
        if (EPI == 1) v = v > 0.f ? v : 0.f;
        if (EPI == 2) v = lrelu2(v);
        C[(size_t)mb * ldc + bn0] = v;
      }
      if (bn1 < N) {
        float v = acc11[r] + bb1;
        if (EPI == 1) v = v > 0.f ? v : 0.f;
        if (EPI == 2) v = lrelu2(v);
        C[(size_t)mb * ldc + bn1] = v;
      }
    }
  }
}

// ---------------- small kernels ----------------
__global__ void fill_u32_k(unsigned* __restrict__ p, unsigned v, long long n) {
  long long i = (long long)blockIdx.x * blockDim.x + threadIdx.x;
  if (i < n) p[i] = v;
}

__global__ void attn_coeff_k(const float* __restrict__ h, const float* __restrict__ asrc,
                             const float* __restrict__ adst, float* __restrict__ als,
                             float* __restrict__ ald) {
  int idx = blockIdx.x * blockDim.x + threadIdx.x;
  if (idx >= NN * HEADS) return;
  int n = idx / HEADS, hd = idx % HEADS;
  const float* hp = h + (size_t)n * HID + hd * FXD;
  const float* as = asrc + hd * FXD;
  const float* ad = adst + hd * FXD;
  float s0 = 0.f, s1 = 0.f;
  for (int c = 0; c < FXD; ++c) { float hv = hp[c]; s0 += hv * as[c]; s1 += hv * ad[c]; }
  als[idx] = s0; ald[idx] = s1;
}

__global__ void edge_max_k(const int* __restrict__ ei, const float* __restrict__ als,
                           const float* __restrict__ ald, unsigned* __restrict__ mkey) {
  long long idx = (long long)blockIdx.x * blockDim.x + threadIdx.x;
  if (idx >= (long long)ETOT * HEADS) return;
  int e = (int)(idx / HEADS), hd = (int)(idx % HEADS);
  int s, d; edge_sd(e, ei, s, d);
  float ee = lrelu2(als[s * HEADS + hd] + ald[d * HEADS + hd]);
  atomicMax(&mkey[d * HEADS + hd], fkey(ee));
}

__global__ void edge_den_deg_k(const int* __restrict__ ei, const float* __restrict__ als,
                               const float* __restrict__ ald, const unsigned* __restrict__ mkey,
                               float* __restrict__ den, float* __restrict__ deg) {
  long long idx = (long long)blockIdx.x * blockDim.x + threadIdx.x;
  if (idx >= (long long)ETOT * HEADS) return;
  int e = (int)(idx / HEADS), hd = (int)(idx % HEADS);
  int s, d; edge_sd(e, ei, s, d);
  float ee = lrelu2(als[s * HEADS + hd] + ald[d * HEADS + hd]);
  unsigned k = mkey[d * HEADS + hd];
  float mv = k ? funkey(k) : 0.f;
  atomicAdd(&den[d * HEADS + hd], __expf(ee - mv));
  if (hd == 0) atomicAdd(&deg[d], 1.0f);
}

__global__ void edge_gat_aggr_k(const int* __restrict__ ei, const float* __restrict__ als,
                                const float* __restrict__ ald, const unsigned* __restrict__ mkey,
                                const float* __restrict__ den, const float* __restrict__ h,
                                float* __restrict__ xg) {
  long long idx = (long long)blockIdx.x * blockDim.x + threadIdx.x;
  if (idx >= (long long)ETOT * HID) return;
  int e = (int)(idx / HID), r = (int)(idx % HID);
  int hd = r / FXD;
  int s, d; edge_sd(e, ei, s, d);
  float ee = lrelu2(als[s * HEADS + hd] + ald[d * HEADS + hd]);
  unsigned k = mkey[d * HEADS + hd];
  float mv = k ? funkey(k) : 0.f;
  float alpha = __expf(ee - mv) / (den[d * HEADS + hd] + 1e-16f);
  atomicAdd(&xg[(size_t)d * HID + r], alpha * h[(size_t)s * HID + r]);
}

__global__ void dinv_k(float* __restrict__ deg) {
  int i = blockIdx.x * blockDim.x + threadIdx.x;
  if (i < NN) deg[i] = rsqrtf(fmaxf(deg[i], 1.0f));
}

__global__ void edge_gcn_aggr_k(const int* __restrict__ ei, const float* __restrict__ dinv,
                                const float* __restrict__ sbuf, float* __restrict__ out) {
  long long idx = (long long)blockIdx.x * blockDim.x + threadIdx.x;
  if (idx >= (long long)ETOT * HID) return;
  int e = (int)(idx / HID), j = (int)(idx % HID);
  int s, d; edge_sd(e, ei, s, d);
  float nrm = dinv[s] * dinv[d];
  atomicAdd(&out[(size_t)d * HID + j], nrm * sbuf[(size_t)s * HID + j]);
}

template<int EPI>
__global__ void bias_act_k(float* __restrict__ x, const float* __restrict__ b, long long n, int ld) {
  long long i = (long long)blockIdx.x * blockDim.x + threadIdx.x;
  if (i >= n) return;
  float v = x[i] + b[(int)(i % ld)];
  if (EPI == 1) v = v > 0.f ? v : 0.f;
  x[i] = v;
}

__global__ void pool_k(const float* __restrict__ xg, const int* __restrict__ batch,
                       unsigned* __restrict__ mxkey, float* __restrict__ sm,
                       float* __restrict__ cnt) {
  long long idx = (long long)blockIdx.x * blockDim.x + threadIdx.x;
  if (idx >= (long long)NN * HID) return;
  int n = (int)(idx / HID), j = (int)(idx % HID);
  int g = batch[n];
  float v = xg[idx];
  atomicMax(&mxkey[g * HID + j], fkey(v));
  atomicAdd(&sm[g * HID + j], v);
  if (j == 0) atomicAdd(&cnt[g], 1.0f);
}

__global__ void xf_k(const unsigned* __restrict__ mxkey, const float* __restrict__ sm,
                     const float* __restrict__ cnt, float* __restrict__ xf) {
  int idx = blockIdx.x * blockDim.x + threadIdx.x;
  if (idx >= NG * HID) return;
  int g = idx / HID, j = idx % HID;
  unsigned k = mxkey[idx];
  float mx = k ? funkey(k) : 0.f;
  if (!isfinite(mx)) mx = 0.f;
  xf[(size_t)g * (2 * HID) + j] = mx;
  xf[(size_t)g * (2 * HID) + HID + j] = sm[idx] / fmaxf(cnt[g], 1.0f);
}

__global__ void dvec_k(const float* __restrict__ A, float* __restrict__ dv) {
  int i = blockIdx.x * blockDim.x + threadIdx.x;
  if (i >= NC) return;
  float s = 0.f;
  for (int j = 0; j < NC; ++j) s += A[i * NC + j];
  dv[i] = rsqrtf(s);
}
__global__ void adj_k(const float* __restrict__ A, const float* __restrict__ dv,
                      float* __restrict__ adj) {
  int idx = blockIdx.x * blockDim.x + threadIdx.x;
  if (idx >= NC * NC) return;
  int i = idx / NC, j = idx % NC;
  adj[idx] = dv[i] * A[j * NC + i] * dv[j];   // ((A@D).T@D)[i,j] = d_i*A[j,i]*d_j
}

// ---------------- host side ----------------
static inline void fill0(hipStream_t st, void* p, long long words) {
  fill_u32_k<<<(unsigned)((words + 255) / 256), 256, 0, st>>>((unsigned*)p, 0u, words);
}

// stage A (f32 [M,K]) and B into padded bf16, then WMMA GEMM.
// b_is_NK: B given row-major as [N,K] (i.e. compute A*B^T), else [K,N].
static inline void gemm(hipStream_t st, const float* A, const float* B, const float* bias,
                        float* C, int M, int N, int K, int lda, int ldb, int ldc,
                        int epi, bool b_is_NK, __bf16* abf, __bf16* btbuf) {
  const int Mp = (M + 31) & ~31, Np = (N + 31) & ~31, Kp = (K + 31) & ~31;
  const long long na = (long long)Mp * Kp, nb = (long long)Np * Kp;
  cvtA_k<<<(unsigned)((na + 255) / 256), 256, 0, st>>>(A, abf, M, K, lda, Kp, na);
  if (b_is_NK) cvtA_k <<<(unsigned)((nb + 255) / 256), 256, 0, st>>>(B, btbuf, N, K, ldb, Kp, nb);
  else         cvtBT_k<<<(unsigned)((nb + 255) / 256), 256, 0, st>>>(B, btbuf, K, N, ldb, Kp, nb);
  dim3 g(Np / 32, (Mp + 127) / 128), blk(128);
  if (epi == 1)      gemm_wmma_bf16<1><<<g, blk, 0, st>>>(abf, btbuf, bias, C, M, N, Kp, ldc);
  else if (epi == 2) gemm_wmma_bf16<2><<<g, blk, 0, st>>>(abf, btbuf, bias, C, M, N, Kp, ldc);
  else               gemm_wmma_bf16<0><<<g, blk, 0, st>>>(abf, btbuf, bias, C, M, N, Kp, ldc);
}

extern "C" void kernel_launch(void* const* d_in, const int* in_sizes, int n_in,
                              void* d_out, int out_size, void* d_ws, size_t ws_size,
                              hipStream_t stream) {
  const float* x     = (const float*)d_in[0];
  const int*   ei    = (const int*)  d_in[1];
  const int*   batch = (const int*)  d_in[2];
  const float* inp   = (const float*)d_in[3];
  const float* Wgat  = (const float*)d_in[4];
  const float* asrc  = (const float*)d_in[5];
  const float* adst  = (const float*)d_in[6];
  const float* bgat  = (const float*)d_in[7];
  const float* Wgcn  = (const float*)d_in[8];
  const float* bgcn  = (const float*)d_in[9];
  const float* Wfc1  = (const float*)d_in[10];
  const float* bfc1  = (const float*)d_in[11];
  const float* Wfc2  = (const float*)d_in[12];
  const float* bfc2  = (const float*)d_in[13];
  const float* Wgc1  = (const float*)d_in[14];
  const float* Wgc2  = (const float*)d_in[15];
  const float* Amat  = (const float*)d_in[16];
  float* out = (float*)d_out;

  char* ws = (char*)d_ws; size_t off = 0;
  auto alloc = [&](size_t bytes) -> char* {
    char* p = ws + off; off += (bytes + 255) & ~(size_t)255; return p;
  };
  float*    hbuf  = (float*)   alloc((size_t)NN * HID * 4);  // x@Wgat, later s=xg@Wgcn
  float*    xg    = (float*)   alloc((size_t)NN * HID * 4);  // GAT out, later GCN scatter target
  float*    als   = (float*)   alloc((size_t)NN * HEADS * 4);
  float*    ald   = (float*)   alloc((size_t)NN * HEADS * 4);
  unsigned* mkey  = (unsigned*)alloc((size_t)NN * HEADS * 4);
  float*    den   = (float*)   alloc((size_t)NN * HEADS * 4);
  float*    deg   = (float*)   alloc((size_t)NN * 4);        // becomes dinv in place
  unsigned* mxkey = (unsigned*)alloc((size_t)NG * HID * 4);
  float*    smbuf = (float*)   alloc((size_t)NG * HID * 4);
  float*    cnt   = (float*)   alloc((size_t)NG * 4);
  float*    xf    = (float*)   alloc((size_t)NG * 2 * HID * 4);
  float*    fc1o  = (float*)   alloc((size_t)NG * FC1 * 4);
  float*    xf2   = (float*)   alloc((size_t)NG * FC2 * 4);
  float*    dv    = (float*)   alloc((size_t)NC * 4);
  float*    adj   = (float*)   alloc((size_t)NC * NC * 4);
  float*    t1    = (float*)   alloc((size_t)NC * 1024 * 4);
  float*    t2    = (float*)   alloc((size_t)NC * 1024 * 4);
  // bf16 staging, reused sequentially by every GEMM (Mp/Np now padded to 32)
  __bf16*   abf   = (__bf16*)  alloc((size_t)30016 * 800 * 2);  // max Mp*Kp
  __bf16*   btbuf = (__bf16*)  alloc((size_t)2400000 * 2);      // max Np*Kp (1504x1568)
  (void)in_sizes; (void)n_in; (void)out_size; (void)ws_size;

  const long long eh = (long long)ETOT * HEADS;
  const long long ec = (long long)ETOT * HID;
  const long long nh = (long long)NN * HID;

  // init accumulators (ws is poisoned once, never re-poisoned between replays)
  fill0(stream, mkey, (long long)NN * HEADS);
  fill0(stream, den,  (long long)NN * HEADS);
  fill0(stream, deg,  NN);
  fill0(stream, xg,   nh);
  fill0(stream, mxkey,(long long)NG * HID);
  fill0(stream, smbuf,(long long)NG * HID);
  fill0(stream, cnt,  NG);

  // --- GAT ---
  gemm(stream, x, Wgat, nullptr, hbuf, NN, HID, FXD, FXD, HID, HID, 0, false, abf, btbuf);
  attn_coeff_k<<<(NN * HEADS + 255) / 256, 256, 0, stream>>>(hbuf, asrc, adst, als, ald);
  edge_max_k<<<(unsigned)((eh + 255) / 256), 256, 0, stream>>>(ei, als, ald, mkey);
  edge_den_deg_k<<<(unsigned)((eh + 255) / 256), 256, 0, stream>>>(ei, als, ald, mkey, den, deg);
  dinv_k<<<(NN + 255) / 256, 256, 0, stream>>>(deg);
  edge_gat_aggr_k<<<(unsigned)((ec + 255) / 256), 256, 0, stream>>>(ei, als, ald, mkey, den, hbuf, xg);
  bias_act_k<1><<<(unsigned)((nh + 255) / 256), 256, 0, stream>>>(xg, bgat, nh, HID);

  // --- GCN --- s = xg@Wgcn into hbuf (hbuf free now)
  gemm(stream, xg, Wgcn, nullptr, hbuf, NN, HID, HID, HID, HID, HID, 0, false, abf, btbuf);
  fill0(stream, xg, nh);
  edge_gcn_aggr_k<<<(unsigned)((ec + 255) / 256), 256, 0, stream>>>(ei, deg, hbuf, xg);
  bias_act_k<1><<<(unsigned)((nh + 255) / 256), 256, 0, stream>>>(xg, bgcn, nh, HID);

  // --- pooling + dense head ---
  pool_k<<<(unsigned)((nh + 255) / 256), 256, 0, stream>>>(xg, batch, mxkey, smbuf, cnt);
  xf_k<<<(NG * HID + 255) / 256, 256, 0, stream>>>(mxkey, smbuf, cnt, xf);
  gemm(stream, xf, Wfc1, bfc1, fc1o, NG, FC1, 2 * HID, 2 * HID, FC1, FC1, 1, false, abf, btbuf);
  gemm(stream, fc1o, Wfc2, bfc2, xf2, NG, FC2, FC1, FC1, FC2, FC2, 0, false, abf, btbuf);

  // --- label GCN branch ---
  dvec_k<<<1, 128, 0, stream>>>(Amat, dv);
  adj_k<<<(NC * NC + 255) / 256, 256, 0, stream>>>(Amat, dv, adj);
  gemm(stream, inp, Wgc1, nullptr, t1, NC, 1024, INCH, INCH, 1024, 1024, 0, false, abf, btbuf);
  gemm(stream, adj, t1, nullptr, t2, NC, 1024, NC, NC, 1024, 1024, 2, false, abf, btbuf);
  gemm(stream, t2, Wgc2, nullptr, t1, NC, 1024, 1024, 1024, 1024, 1024, 0, false, abf, btbuf);
  gemm(stream, adj, t1, nullptr, t2, NC, 1024, NC, NC, 1024, 1024, 0, false, abf, btbuf);

  // --- out = xf2 @ y^T (t2 is y, row-major [N,K]) ---
  gemm(stream, xf2, t2, nullptr, out, NG, NC, 1024, 1024, 1024, NC, 0, true, abf, btbuf);
}
